// selfattention_89747636617288
// MI455X (gfx1250) — matrix operand order repaired and checked by reading, not compiled, not run on hardware
//
#include <hip/hip_runtime.h>
#include <hip/hip_bf16.h>

#define EMBED  512
#define HEADS  8
#define HDIM   64
#define NBATCH 4
#define SEQ    2048

typedef __attribute__((ext_vector_type(16))) _Float16 v16h;
typedef __attribute__((ext_vector_type(8)))  _Float16 v8h;
typedef __attribute__((ext_vector_type(8)))  float    v8f;

__device__ __forceinline__ v8f wmma16(v16h a, v16h b, v8f c) {
  // D = A(16x32 f16) * B(32x16 f16) + C(16x16 f32)
  return __builtin_amdgcn_wmma_f32_16x16x32_f16(false, a, false, b, (short)0, c,
                                                false, false);
}

__device__ __forceinline__ v16h cat8(v8h a, v8h b) {
  return __builtin_shufflevector(a, b, 0, 1, 2, 3, 4, 5, 6, 7, 8, 9, 10, 11, 12,
                                 13, 14, 15);
}

// reduce across the 16-lane half-wave (xor masks 1..8 stay within halves)
__device__ __forceinline__ float hmax16(float v) {
#pragma unroll
  for (int off = 8; off >= 1; off >>= 1) v = fmaxf(v, __shfl_xor(v, off, 32));
  return v;
}
__device__ __forceinline__ float hsum16(float v) {
#pragma unroll
  for (int off = 8; off >= 1; off >>= 1) v += __shfl_xor(v, off, 32);
  return v;
}

// Async global -> LDS copy of one 16-byte chunk per lane (CDNA5 TDM-lite path).
__device__ __forceinline__ void async_copy_b128(const _Float16* gptr, _Float16* lptr) {
  unsigned lds_off = (unsigned)(size_t)lptr;        // low 32 bits = LDS offset
  unsigned long long gaddr = (unsigned long long)(size_t)gptr;
  asm volatile("global_load_async_to_lds_b128 %0, %1, off"
               :: "v"(lds_off), "v"(gaddr)
               : "memory");
}
__device__ __forceinline__ void wait_async0() {
  asm volatile("s_wait_asynccnt 0x0" ::: "memory");
}

// Build WMMA A-matrix (16x32 f16) chunk c from a row of fp32 (converted on the fly).
// A layout: lane -> row m=lane&15; halfs j<8 -> K=32c+8*hi+j ; j>=8 -> K=32c+16+8*hi+(j-8)
__device__ __forceinline__ v16h a_from_f32(const float* rowptr, int c, int hi) {
  const float* p0 = rowptr + 32 * c + 8 * hi;
  const float* p1 = rowptr + 32 * c + 16 + 8 * hi;
  v16h a;
#pragma unroll
  for (int j = 0; j < 8; ++j) {
    a[j]     = (_Float16)p0[j];
    a[8 + j] = (_Float16)p1[j];
  }
  return a;
}

// -------------------- kernel 1: weight fp32 -> f16 --------------------
__global__ void wconv_kernel(const float* __restrict__ Wq,
                             const float* __restrict__ Wk,
                             const float* __restrict__ Wv,
                             const float* __restrict__ Wo,
                             _Float16* __restrict__ wqh, _Float16* __restrict__ wkh,
                             _Float16* __restrict__ wvh, _Float16* __restrict__ woh) {
  int i = blockIdx.x * blockDim.x + threadIdx.x;
  if (i < HDIM * HDIM) {
    wqh[i] = (_Float16)Wq[i];
    wkh[i] = (_Float16)Wk[i];
    wvh[i] = (_Float16)Wv[i];
  }
  if (i < EMBED * EMBED) woh[i] = (_Float16)Wo[i];
}

// -------------------- kernel 2: fused per-head QKV projection --------------------
// Y(16x64) = X(16x64) @ W^T(64x64) : B[k=d][n=e] = W[e][d] (contiguous in d per lane)
__device__ __forceinline__ void project_store(const float* rowptr,
                                              const _Float16* __restrict__ w,
                                              _Float16* __restrict__ outRM,
                                              _Float16* __restrict__ outT, int lt,
                                              int lo, int hi) {
  v16h a0 = a_from_f32(rowptr, 0, hi);
  v16h a1 = a_from_f32(rowptr, 1, hi);
  v8f zero = {};
#pragma unroll
  for (int t = 0; t < 4; ++t) {
    const _Float16* wrow = w + (size_t)(16 * t + lo) * HDIM;
    v16h b0 = cat8(*(const v8h*)(wrow + 8 * hi), *(const v8h*)(wrow + 16 + 8 * hi));           // K chunk 0
    v16h b1 = cat8(*(const v8h*)(wrow + 32 + 8 * hi), *(const v8h*)(wrow + 48 + 8 * hi));      // K chunk 1
    v8f c = wmma16(a1, b1, wmma16(a0, b0, zero));
    if (outRM) {
#pragma unroll
      for (int r = 0; r < 8; ++r)
        outRM[(size_t)(lt * 16 + 8 * hi + r) * HDIM + 16 * t + lo] = (_Float16)c[r];
    }
    if (outT) {  // transposed [d][l]: 8 consecutive rows -> one b128 store
      v8h pk;
#pragma unroll
      for (int r = 0; r < 8; ++r) pk[r] = (_Float16)c[r];
      *(v8h*)(outT + (size_t)(16 * t + lo) * SEQ + lt * 16 + 8 * hi) = pk;
    }
  }
}

__global__ __launch_bounds__(128) void qkv_kernel(
    const float* __restrict__ Qin, const float* __restrict__ Kin,
    const float* __restrict__ Vin, const _Float16* __restrict__ wq,
    const _Float16* __restrict__ wk, const _Float16* __restrict__ wv,
    _Float16* __restrict__ Qp, _Float16* __restrict__ Kp,
    _Float16* __restrict__ Vt) {
  const int lane = threadIdx.x & 31, wave = threadIdx.x >> 5;
  const int lo = lane & 15, hi = lane >> 4;
  const int lt = blockIdx.x * 4 + wave;  // 16-row tile index in sequence
  const int h = blockIdx.y, n = blockIdx.z;
  const size_t rowbase = ((size_t)n * SEQ + lt * 16 + lo) * EMBED + h * HDIM;
  _Float16* qdst = Qp + (size_t)(n * HEADS + h) * SEQ * HDIM;
  _Float16* kdst = Kp + (size_t)(n * HEADS + h) * SEQ * HDIM;
  _Float16* vdst = Vt + (size_t)(n * HEADS + h) * HDIM * SEQ;
  project_store(Qin + rowbase, wq, qdst, nullptr, lt, lo, hi);
  project_store(Kin + rowbase, wk, kdst, nullptr, lt, lo, hi);
  project_store(Vin + rowbase, wv, nullptr, vdst, lt, lo, hi);
}

// -------------------- kernel 3: flash attention per (n,h,q-tile) --------------------
// 4 waves/block share (n,h); K/V 32-key tiles are staged once per block into LDS
// via async global->LDS copies, double-buffered to overlap with WMMA compute.
__global__ __launch_bounds__(128) void attn_kernel(
    const _Float16* __restrict__ Qp, const _Float16* __restrict__ Kp,
    const _Float16* __restrict__ Vt, const int* __restrict__ mask,
    _Float16* __restrict__ AO) {
  __shared__ __align__(64) _Float16 pbuf[4][16 * 32];
  __shared__ __align__(64) _Float16 kbuf[2][32 * 64];  // [key_local][d]
  __shared__ __align__(64) _Float16 vbuf[2][64 * 32];  // [d][key_local]
  const int tid = threadIdx.x;
  const int lane = tid & 31, wave = tid >> 5;
  const int lo = lane & 15, hi = lane >> 4;
  const int qt = blockIdx.x * 4 + wave;
  const int h = blockIdx.y, n = blockIdx.z;
  const size_t hoff = (size_t)(n * HEADS + h) * SEQ * HDIM;
  const _Float16* qpb = Qp + hoff;
  const _Float16* kpb = Kp + hoff;
  const _Float16* vtb = Vt + (size_t)(n * HEADS + h) * HDIM * SEQ;
  const int* mrow = mask + (size_t)n * SEQ * SEQ + (size_t)(qt * 16) * SEQ;
  _Float16* pb = pbuf[wave];

  // stage one 32-key K tile (32x64 f16) + V tile (64x32 f16) into LDS buffer
  auto stage = [&](int kb, int buf) {
#pragma unroll
    for (int i = 0; i < 2; ++i) {  // K: 256 chunks of 16B, 8 chunks per key row
      int chunk = tid + i * 128;
      int row = chunk >> 3, c8 = chunk & 7;
      async_copy_b128(kpb + (size_t)(kb + row) * HDIM + c8 * 8,
                      kbuf[buf] + row * 64 + c8 * 8);
    }
#pragma unroll
    for (int i = 0; i < 2; ++i) {  // V: 256 chunks of 16B, 4 chunks per d row
      int chunk = tid + i * 128;
      int row = chunk >> 2, c8 = chunk & 3;
      async_copy_b128(vtb + (size_t)row * SEQ + kb + c8 * 8,
                      vbuf[buf] + row * 32 + c8 * 8);
    }
  };

  // Q tile as two 16x32 A chunks (resident for whole loop)
  const _Float16* qrow = qpb + (size_t)(qt * 16 + lo) * HDIM;
  v16h aq0 = cat8(*(const v8h*)(qrow + 8 * hi), *(const v8h*)(qrow + 16 + 8 * hi));
  v16h aq1 = cat8(*(const v8h*)(qrow + 32 + 8 * hi),
                  *(const v8h*)(qrow + 48 + 8 * hi));

  float mrun[8], lrun[8];
  v8f acc[4];
  v8f zero = {};
#pragma unroll
  for (int r = 0; r < 8; ++r) {
    mrun[r] = -3.0e38f;
    lrun[r] = 0.0f;
  }
#pragma unroll
  for (int t = 0; t < 4; ++t) acc[t] = zero;

  const float scale = 0.04419417382415922f;  // 1/sqrt(512)
  const float maskval = -4.419417382e18f;    // -1e20 * scale

  stage(0, 0);      // prologue: fill buffer 0
  wait_async0();
  __syncthreads();

  for (int kb = 0; kb < SEQ; kb += 32) {
    const int buf = (kb >> 5) & 1;
    if (kb + 32 < SEQ) stage(kb + 32, buf ^ 1);  // overlap next-tile copy

    const _Float16* kl = kbuf[buf];
    const _Float16* vl = vbuf[buf];

    // energy: two 16x16 tiles over 32 keys, K-dim = 64 (B operands from LDS)
    v8f e[2];
#pragma unroll
    for (int t = 0; t < 2; ++t) {
      const _Float16* krow = kl + (16 * t + lo) * 64;
      v16h b0 = cat8(*(const v8h*)(krow + 8 * hi), *(const v8h*)(krow + 16 + 8 * hi));
      v16h b1 = cat8(*(const v8h*)(krow + 32 + 8 * hi), *(const v8h*)(krow + 48 + 8 * hi));
      e[t] = wmma16(aq1, b1, wmma16(aq0, b0, zero));
    }
    // mask (pre-scale -1e20 semantics folded in) + scale
#pragma unroll
    for (int t = 0; t < 2; ++t)
#pragma unroll
      for (int r = 0; r < 8; ++r) {
        int m = 8 * hi + r;
        int k = kb + 16 * t + lo;
        float ev = e[t][r] * scale;
        if (mrow[(size_t)m * SEQ + k] == 0) ev = maskval;
        e[t][r] = ev;
      }
    // online softmax update
#pragma unroll
    for (int r = 0; r < 8; ++r) {
      float rm = hmax16(fmaxf(e[0][r], e[1][r]));
      float mN = fmaxf(mrun[r], rm);
      float corr = __expf(mrun[r] - mN);
      mrun[r] = mN;
      lrun[r] *= corr;
#pragma unroll
      for (int t = 0; t < 4; ++t) acc[t][r] *= corr;
    }
#pragma unroll
    for (int r = 0; r < 8; ++r) {
      float p0 = __expf(e[0][r] - mrun[r]);
      float p1 = __expf(e[1][r] - mrun[r]);
      e[0][r] = p0;
      e[1][r] = p1;
      lrun[r] += hsum16(p0 + p1);
    }
    // re-layout P (16x32) through per-wave LDS into WMMA A layout
#pragma unroll
    for (int t = 0; t < 2; ++t)
#pragma unroll
      for (int r = 0; r < 8; ++r)
        pb[(8 * hi + r) * 32 + 16 * t + lo] = (_Float16)e[t][r];
    v16h pa = cat8(*(const v8h*)(pb + lo * 32 + 8 * hi),
                   *(const v8h*)(pb + lo * 32 + 16 + 8 * hi));
    // PV: acc(16x64) += P(16x32) * V(32x64)  (B operands from LDS V tile)
#pragma unroll
    for (int t = 0; t < 4; ++t) {
      const _Float16* vrowp = vl + (16 * t + lo) * 32;
      acc[t] = wmma16(pa, cat8(*(const v8h*)(vrowp + 8 * hi), *(const v8h*)(vrowp + 16 + 8 * hi)), acc[t]);
    }

    wait_async0();    // next tile landed in LDS
    __syncthreads();  // publish across the 4 waves; buffer reuse safe
  }

  // normalize + store (f16) to attention-output activation [n*S+q][h*64+d]
  _Float16* aor = AO + (size_t)(n * SEQ + qt * 16) * EMBED + h * HDIM;
#pragma unroll
  for (int t = 0; t < 4; ++t)
#pragma unroll
    for (int r = 0; r < 8; ++r) {
      float o = acc[t][r] / lrun[r];
      aor[(size_t)(8 * hi + r) * EMBED + 16 * t + lo] = (_Float16)o;
    }
}

// -------------------- kernel 4: output projection + bias --------------------
__global__ __launch_bounds__(128) void oproj_kernel(const _Float16* __restrict__ AO,
                                                    const _Float16* __restrict__ wo,
                                                    const float* __restrict__ bo,
                                                    float* __restrict__ out) {
  const int lane = threadIdx.x & 31, wave = threadIdx.x >> 5;
  const int lo = lane & 15, hi = lane >> 4;
  const int rt = blockIdx.x * 4 + wave;  // 16-row tile of 8192 rows
  const int et = blockIdx.y;             // 64-col tile of 512 cols
  const _Float16* arow = AO + (size_t)(rt * 16 + lo) * EMBED;
  v8f zero = {};
  v8f c[4] = {zero, zero, zero, zero};
#pragma unroll
  for (int kc = 0; kc < 16; ++kc) {  // K = 512 in chunks of 32
    v16h a = cat8(*(const v8h*)(arow + 32 * kc + 8 * hi),
                  *(const v8h*)(arow + 32 * kc + 16 + 8 * hi));
#pragma unroll
    for (int t = 0; t < 4; ++t) {
      const _Float16* wrow =
          wo + (size_t)(et * 64 + 16 * t + lo) * EMBED + 32 * kc;
      c[t] = wmma16(a, cat8(*(const v8h*)(wrow + 8 * hi), *(const v8h*)(wrow + 16 + 8 * hi)), c[t]);
    }
  }
#pragma unroll
  for (int t = 0; t < 4; ++t) {
    int e = et * 64 + 16 * t + lo;
    float bias = bo[e];
#pragma unroll
    for (int r = 0; r < 8; ++r)
      out[(size_t)(rt * 16 + 8 * hi + r) * EMBED + e] = c[t][r] + bias;
  }
}

extern "C" void kernel_launch(void* const* d_in, const int* in_sizes, int n_in,
                              void* d_out, int out_size, void* d_ws, size_t ws_size,
                              hipStream_t stream) {
  const float* value = (const float*)d_in[0];
  const float* key_i = (const float*)d_in[1];
  const float* query = (const float*)d_in[2];
  const int* mask = (const int*)d_in[3];
  const float* Wq = (const float*)d_in[4];
  const float* Wk = (const float*)d_in[5];
  const float* Wv = (const float*)d_in[6];
  const float* Wo = (const float*)d_in[7];
  const float* bo = (const float*)d_in[8];
  float* out = (float*)d_out;

  _Float16* ws = (_Float16*)d_ws;
  const size_t PROJ = (size_t)NBATCH * HEADS * SEQ * HDIM;  // 4M halfs each
  _Float16* Qp = ws;
  _Float16* Kp = Qp + PROJ;
  _Float16* Vt = Kp + PROJ;
  _Float16* AO = Vt + PROJ;                    // N*S x 512
  _Float16* wqh = AO + (size_t)NBATCH * SEQ * EMBED;
  _Float16* wkh = wqh + HDIM * HDIM;
  _Float16* wvh = wkh + HDIM * HDIM;
  _Float16* woh = wvh + HDIM * HDIM;

  wconv_kernel<<<(EMBED * EMBED + 255) / 256, 256, 0, stream>>>(
      Wq, Wk, Wv, Wo, wqh, wkh, wvh, woh);
  qkv_kernel<<<dim3(SEQ / 64, HEADS, NBATCH), 128, 0, stream>>>(
      query, key_i, value, wqh, wkh, wvh, Qp, Kp, Vt);
  attn_kernel<<<dim3(SEQ / 64, HEADS, NBATCH), 128, 0, stream>>>(Qp, Kp, Vt, mask,
                                                                 AO);
  oproj_kernel<<<dim3(NBATCH * SEQ / 64, EMBED / 64), 128, 0, stream>>>(AO, woh, bo,
                                                                        out);
}
